// SoftMoE5_3719441678653
// MI455X (gfx1250) — compile-verified, matrix operand order checked by reading, not run on hardware
//
#include <hip/hip_runtime.h>
#include <hip/hip_bf16.h>
#include <math.h>

typedef __attribute__((ext_vector_type(16))) _Float16 v16h;
typedef __attribute__((ext_vector_type(8)))  _Float16 v8h;
typedef __attribute__((ext_vector_type(8)))  float    v8f;
typedef __attribute__((ext_vector_type(4)))  int      v4i;
typedef __attribute__((ext_vector_type(8)))  int      v8i;
typedef __attribute__((ext_vector_type(4)))  unsigned int v4u;

#define B_  8
#define N_  2048
#define D_  2048
#define E_  16
#define S_  128
#define H_  512
#define ES_ 2048
#define BN_ 16384

#define WMMA_F16(a, b, c) \
    __builtin_amdgcn_wmma_f32_16x16x32_f16(false, (a), false, (b), (short)0, (c), false, false)

// ---- CDNA5 TDM (tensor DMA) path ----
#if __has_builtin(__builtin_amdgcn_tensor_load_to_lds) && __has_builtin(__builtin_amdgcn_s_wait_tensorcnt)
#define HAVE_TDM 1
#endif

// ---- CDNA5 async global->LDS path ----
#if __has_builtin(__builtin_amdgcn_global_load_async_to_lds_b128)
#define HAVE_ASYNC_LDS 1
#define ASYNC_B128(gsrc, ldst) \
    __builtin_amdgcn_global_load_async_to_lds_b128( \
        (__attribute__((address_space(1))) v4i*)(gsrc), \
        (__attribute__((address_space(3))) v4i*)(ldst), 0, 0)
#endif

#if __has_builtin(__builtin_amdgcn_s_wait_asynccnt)
#define WAIT_ASYNC(n) __builtin_amdgcn_s_wait_asynccnt(n)
#else
#define WAIT_ASYNC(n) asm volatile("s_wait_asynccnt %0" :: "n"(n) : "memory")
#endif

__device__ __forceinline__ unsigned lds_addr_of(const void* p) {
    return (unsigned)(unsigned long long)(__attribute__((address_space(3))) const void*)p;
}

#ifdef HAVE_TDM
// 2D f16 tile load via Tensor Data Mover: tile_w x tile_h elements, row stride in elements,
// LDS padding 8 DWORDs after every 32 DWORDs (row 128B -> LDS row stride 160B = 80 f16).
__device__ __forceinline__ void tdm_load_2d_f16(
    const void* gaddr, unsigned lds_off, int tile_w, int tile_h, long long row_stride)
{
    unsigned long long ga = (unsigned long long)gaddr;
    v4u g0;
    g0[0] = 1u;                                              // count=1 (valid user D#)
    g0[1] = lds_off;                                         // lds_addr (bytes)
    g0[2] = (unsigned)(ga & 0xFFFFFFFFu);                    // global_addr[31:0]
    g0[3] = (unsigned)((ga >> 32) & 0x1FFFFFFu) | (2u << 30); // global_addr[56:32] | type=2
    v8i g1;
    g1[0] = (1 << 16)            // data_size = 1 -> 2 bytes
          | (1 << 20)            // pad_enable
          | (4 << 22)            // pad_interval: 32 DWORDs (128B) of data...
          | (7 << 25);           // pad_amount:   ...then 8 DWORDs (32B) padding
    g1[1] = (tile_w & 0xFFFF) << 16;                         // tensor_dim0[15:0] (bits 63:48)
    g1[2] = ((unsigned)tile_w >> 16) | ((tile_h & 0xFFFF) << 16); // dim0[31:16] | dim1[15:0]
    g1[3] = ((unsigned)tile_h >> 16) | (tile_w << 16);       // dim1[31:16] | tile_dim0
    g1[4] = tile_h & 0xFFFF;                                 // tile_dim1 (tile_dim2 = 0)
    g1[5] = (int)(row_stride & 0xFFFFFFFF);                  // tensor_dim0_stride[31:0]
    g1[6] = (int)((row_stride >> 32) & 0xFFFF);              // stride[47:32] (dim1_stride unused, 2D)
    g1[7] = 0;
    v4i z4 = {0, 0, 0, 0};                                   // groups 2/3: unused for 2D tiles
    v8i z8 = {0, 0, 0, 0, 0, 0, 0, 0};                       // extra group (6-arg variant): unused
    __builtin_amdgcn_tensor_load_to_lds(g0, g1, z4, z4, z8, 0);
}
#endif

// ---------------- block reduction helpers ----------------
__device__ __forceinline__ float block_reduce_sum(float v, float* red) {
    int tid = threadIdx.x;
    red[tid] = v; __syncthreads();
    for (int o = 128; o > 0; o >>= 1) {
        if (tid < o) red[tid] += red[tid + o];
        __syncthreads();
    }
    float r = red[0]; __syncthreads();
    return r;
}
__device__ __forceinline__ float block_reduce_max(float v, float* red) {
    int tid = threadIdx.x;
    red[tid] = v; __syncthreads();
    for (int o = 128; o > 0; o >>= 1) {
        if (tid < o) red[tid] = fmaxf(red[tid], red[tid + o]);
        __syncthreads();
    }
    float r = red[0]; __syncthreads();
    return r;
}

// ---------------- LayerNorm + l2norm (per token) ----------------
__global__ __launch_bounds__(256) void ln_kernel(
    const float* __restrict__ x, const float* __restrict__ gamma,
    const float* __restrict__ beta, const float* __restrict__ scale,
    _Float16* __restrict__ xln, _Float16* __restrict__ xn)
{
    __shared__ float red[256];
    size_t row = blockIdx.x;
    const float* xr = x + row * D_;
    int tid = threadIdx.x;
    float v[8];
    float s = 0.f;
    #pragma unroll
    for (int i = 0; i < 8; ++i) { v[i] = xr[tid + i * 256]; s += v[i]; }
    float mean = block_reduce_sum(s, red) * (1.0f / D_);
    s = 0.f;
    #pragma unroll
    for (int i = 0; i < 8; ++i) { float d = v[i] - mean; s += d * d; }
    float var = block_reduce_sum(s, red) * (1.0f / D_);
    float rstd = rsqrtf(var + 1e-5f);
    s = 0.f;
    #pragma unroll
    for (int i = 0; i < 8; ++i) {
        int d = tid + i * 256;
        float xl = (v[i] - mean) * rstd * gamma[d] + beta[d];
        v[i] = xl; s += xl * xl;
    }
    float nrm = sqrtf(block_reduce_sum(s, red));
    float inv = scale[0] / fmaxf(nrm, 1e-12f);
    #pragma unroll
    for (int i = 0; i < 8; ++i) {
        int d = tid + i * 256;
        xln[row * D_ + d] = (_Float16)v[i];
        xn [row * D_ + d] = (_Float16)(v[i] * inv);
    }
}

// ---------------- mu l2norm over D, write mu_n^T [ES][D] ----------------
__global__ __launch_bounds__(256) void mu_norm_kernel(
    const float* __restrict__ mu, _Float16* __restrict__ munT)
{
    int es = blockIdx.x * 256 + threadIdx.x;
    float s = 0.f;
    for (int d = 0; d < D_; ++d) {
        float m = mu[(size_t)d * ES_ + es];
        s += m * m;
    }
    float inv = 1.0f / fmaxf(sqrtf(s), 1e-12f);
    for (int d = 0; d < D_; ++d)
        munT[(size_t)es * D_ + d] = (_Float16)(mu[(size_t)d * ES_ + es] * inv);
}

// ---------------- softmax over tokens N (dispatch), transposed + LDS-staged writes ----------------
__global__ __launch_bounds__(256) void dispatch_softmax_kernel(
    const float* __restrict__ logits, _Float16* __restrict__ dispT)
{
    __shared__ _Float16 tile[256][33];
    int tid = threadIdx.x;
    int es0 = blockIdx.x * 256;
    int es  = es0 + tid;
    int b   = blockIdx.y;
    const float* base = logits + (size_t)b * N_ * ES_ + es;

    float mx = -3.0e38f;
    for (int n = 0; n < N_; ++n) mx = fmaxf(mx, base[(size_t)n * ES_]);
    float s = 0.f;
    for (int n = 0; n < N_; ++n) s += expf(base[(size_t)n * ES_] - mx);
    float inv = 1.0f / s;

    _Float16* dbase = dispT + (size_t)b * ES_ * N_;
    for (int nt = 0; nt < N_; nt += 32) {
        #pragma unroll
        for (int i = 0; i < 32; ++i)
            tile[tid][i] = (_Float16)(expf(base[(size_t)(nt + i) * ES_] - mx) * inv);
        __syncthreads();
        #pragma unroll
        for (int i = 0; i < 32; ++i) {
            int r = i * 8 + (tid >> 5);
            int c = tid & 31;
            dbase[(size_t)(es0 + r) * N_ + nt + c] = tile[r][c];
        }
        __syncthreads();
    }
}

// ---------------- softmax over slots ES (combine), row-major ----------------
__global__ __launch_bounds__(256) void combine_softmax_kernel(
    const float* __restrict__ logits, _Float16* __restrict__ comb)
{
    __shared__ float red[256];
    size_t row = blockIdx.x;
    const float* lr = logits + row * ES_;
    int tid = threadIdx.x;
    float v[8];
    float mx = -3.0e38f;
    #pragma unroll
    for (int i = 0; i < 8; ++i) { v[i] = lr[tid + i * 256]; mx = fmaxf(mx, v[i]); }
    mx = block_reduce_max(mx, red);
    float s = 0.f;
    #pragma unroll
    for (int i = 0; i < 8; ++i) { v[i] = expf(v[i] - mx); s += v[i]; }
    s = block_reduce_sum(s, red);
    float inv = 1.0f / s;
    #pragma unroll
    for (int i = 0; i < 8; ++i)
        comb[row * ES_ + tid + i * 256] = (_Float16)(v[i] * inv);
}

// ---------------- batched transpose (fp32 or f16 in) -> f16 out ----------------
template <typename T>
__global__ __launch_bounds__(256) void transpose_to_f16(
    const T* __restrict__ in, _Float16* __restrict__ out, int R, int Cc)
{
    __shared__ _Float16 tile[32][33];
    size_t zoff = (size_t)blockIdx.z * R * Cc;
    int x = blockIdx.x * 32 + threadIdx.x;
    int y0 = blockIdx.y * 32;
    #pragma unroll
    for (int j = 0; j < 32; j += 8)
        tile[threadIdx.y + j][threadIdx.x] =
            (_Float16)in[zoff + (size_t)(y0 + threadIdx.y + j) * Cc + x];
    __syncthreads();
    int ox  = blockIdx.y * 32 + threadIdx.x;
    int oy0 = blockIdx.x * 32;
    #pragma unroll
    for (int j = 0; j < 32; j += 8)
        out[zoff + (size_t)(oy0 + threadIdx.y + j) * R + ox] = tile[threadIdx.x][threadIdx.y + j];
}

// ---------------- LDS-staged WMMA f16 GEMM: C(MxN) = A(MxK) * B^T(NxK) ----------------
// Block tile 128x128, BK=64; 8 waves (4 M x 2 N), each wave 32x64 (2x4 WMMA frags).
// Staging: TDM tensor DMA (TENSORcnt) > per-lane async-to-LDS (ASYNCcnt) > sync loads.
// EPI: 0 = fp32 store, 1 = f16 store, 2 = bias+GELU f16 store,
//      3 = bias + transposed f16 store, 4 = fp32 non-temporal store
template <int EPI>
__global__ __launch_bounds__(256) void gemm_wmma_f16(
    const _Float16* __restrict__ A, const _Float16* __restrict__ Bt,
    void* __restrict__ C, const float* __restrict__ bias,
    int K, int lda, int ldb, int ldc,
    long long Abs, long long Bbs, int bmod,
    long long Cbs0, long long Cbs1, int czdiv, long long bias_bs)
{
    constexpr int LDSP = 80; // padded LDS row stride (f16): 160B rows, 32B-aligned frags
#if defined(HAVE_TDM) || defined(HAVE_ASYNC_LDS)
    constexpr int NBUF = 2;
#else
    constexpr int NBUF = 1;
#endif
    constexpr int BUFO = 128 * LDSP;
    __shared__ alignas(32) _Float16 As[NBUF * BUFO];
    __shared__ alignas(32) _Float16 Bs[NBUF * BUFO];

    int z = blockIdx.z;
    const _Float16* Ab = A  + (size_t)z * Abs + (size_t)blockIdx.y * 128 * lda;
    const _Float16* Bb = Bt + (size_t)(z % bmod) * Bbs + (size_t)blockIdx.x * 128 * ldb;
    size_t Coff = (size_t)(z / czdiv) * Cbs0 + (size_t)(z % czdiv) * Cbs1;
    const float* biasb = (EPI == 2 || EPI == 3) ? (bias + (size_t)(z % bmod) * bias_bs)
                                                : (const float*)nullptr;

    int t    = threadIdx.x;
    int lane = t & 31;
    int w    = t >> 5;
    int wm   = w & 3;              // 4 waves along M, 32 rows each
    int wn   = w >> 2;             // 2 waves along N, 64 cols each
    int lr   = lane & 15;
    int kqh  = (lane >> 4) * 16;   // K-half per lane half: contiguous k = 16h + s, same for A and B

    v8f acc[2][4] = {};

#ifdef HAVE_TDM
    // ---- TDM-staged, double-buffered: one tensor DMA per operand tile, issued by wave 0 ----
    unsigned asA0 = lds_addr_of(As), asA1 = lds_addr_of(As + BUFO);
    unsigned asB0 = lds_addr_of(Bs), asB1 = lds_addr_of(Bs + BUFO);
    int ntile = K >> 6;
    if (w == 0) {
        tdm_load_2d_f16(Ab, asA0, 64, 128, lda);
        tdm_load_2d_f16(Bb, asB0, 64, 128, ldb);
    }
    for (int ti = 0; ti < ntile; ++ti) {
        int cur = ti & 1;
        __syncthreads();                       // all waves done reading buf[cur^1]
        if (w == 0) {
            if (ti + 1 < ntile) {
                tdm_load_2d_f16(Ab + (ti + 1) * 64, cur ? asA0 : asA1, 64, 128, lda);
                tdm_load_2d_f16(Bb + (ti + 1) * 64, cur ? asB0 : asB1, 64, 128, ldb);
                __builtin_amdgcn_s_wait_tensorcnt(2);  // tile ti done; ti+1 in flight
            } else {
                __builtin_amdgcn_s_wait_tensorcnt(0);
            }
        }
        __syncthreads();                       // wave 0's wait gates everyone
        const _Float16* Asr = As + cur * BUFO;
        const _Float16* Bsr = Bs + cur * BUFO;
        #pragma unroll
        for (int ks = 0; ks < 64; ks += 32) {
            int kb = ks + kqh;
            v16h a0  = *(const v16h*)(Asr + (wm * 32      + lr) * LDSP + kb);
            v16h a1  = *(const v16h*)(Asr + (wm * 32 + 16 + lr) * LDSP + kb);
            v16h b0  = *(const v16h*)(Bsr + (wn * 64      + lr) * LDSP + kb);
            v16h b1  = *(const v16h*)(Bsr + (wn * 64 + 16 + lr) * LDSP + kb);
            v16h b2v = *(const v16h*)(Bsr + (wn * 64 + 32 + lr) * LDSP + kb);
            v16h b3v = *(const v16h*)(Bsr + (wn * 64 + 48 + lr) * LDSP + kb);
            acc[0][0] = WMMA_F16(a0, b0,  acc[0][0]);
            acc[0][1] = WMMA_F16(a0, b1,  acc[0][1]);
            acc[0][2] = WMMA_F16(a0, b2v, acc[0][2]);
            acc[0][3] = WMMA_F16(a0, b3v, acc[0][3]);
            acc[1][0] = WMMA_F16(a1, b0,  acc[1][0]);
            acc[1][1] = WMMA_F16(a1, b1,  acc[1][1]);
            acc[1][2] = WMMA_F16(a1, b2v, acc[1][2]);
            acc[1][3] = WMMA_F16(a1, b3v, acc[1][3]);
        }
    }
#else
    // cooperative tile load mapping: thread -> (row, 16B column chunk), 4 row-groups
    int lrow = t >> 3;             // 0..31
    int lcol = (t & 7) * 8;        // 0,8,..,56
    const _Float16* Ag = Ab + (size_t)lrow * lda + lcol;
    const _Float16* Bg = Bb + (size_t)lrow * ldb + lcol;
    _Float16* Asw = As + lrow * LDSP + lcol;
    _Float16* Bsw = Bs + lrow * LDSP + lcol;
#ifdef HAVE_ASYNC_LDS
    #pragma unroll
    for (int i = 0; i < 4; ++i) {
        ASYNC_B128(Ag + (size_t)(32 * i) * lda, Asw + 32 * i * LDSP);
        ASYNC_B128(Bg + (size_t)(32 * i) * ldb, Bsw + 32 * i * LDSP);
    }
    const _Float16* agn = Ag + 64;
    const _Float16* bgn = Bg + 64;
    int ntile = K >> 6;
    for (int ti = 0; ti < ntile; ++ti) {
        int cur = ti & 1;
        __syncthreads();
        if (ti + 1 < ntile) {
            int nxt = cur ^ 1;
            #pragma unroll
            for (int i = 0; i < 4; ++i) {
                ASYNC_B128(agn + (size_t)(32 * i) * lda, Asw + nxt * BUFO + 32 * i * LDSP);
                ASYNC_B128(bgn + (size_t)(32 * i) * ldb, Bsw + nxt * BUFO + 32 * i * LDSP);
            }
            agn += 64; bgn += 64;
            WAIT_ASYNC(8);
        } else {
            WAIT_ASYNC(0);
        }
        __syncthreads();
        const _Float16* Asr = As + cur * BUFO;
        const _Float16* Bsr = Bs + cur * BUFO;
        #pragma unroll
        for (int ks = 0; ks < 64; ks += 32) {
            int kb = ks + kqh;
            v16h a0  = *(const v16h*)(Asr + (wm * 32      + lr) * LDSP + kb);
            v16h a1  = *(const v16h*)(Asr + (wm * 32 + 16 + lr) * LDSP + kb);
            v16h b0  = *(const v16h*)(Bsr + (wn * 64      + lr) * LDSP + kb);
            v16h b1  = *(const v16h*)(Bsr + (wn * 64 + 16 + lr) * LDSP + kb);
            v16h b2v = *(const v16h*)(Bsr + (wn * 64 + 32 + lr) * LDSP + kb);
            v16h b3v = *(const v16h*)(Bsr + (wn * 64 + 48 + lr) * LDSP + kb);
            acc[0][0] = WMMA_F16(a0, b0,  acc[0][0]);
            acc[0][1] = WMMA_F16(a0, b1,  acc[0][1]);
            acc[0][2] = WMMA_F16(a0, b2v, acc[0][2]);
            acc[0][3] = WMMA_F16(a0, b3v, acc[0][3]);
            acc[1][0] = WMMA_F16(a1, b0,  acc[1][0]);
            acc[1][1] = WMMA_F16(a1, b1,  acc[1][1]);
            acc[1][2] = WMMA_F16(a1, b2v, acc[1][2]);
            acc[1][3] = WMMA_F16(a1, b3v, acc[1][3]);
        }
    }
#else
    for (int kt = 0; kt < K; kt += 64) {
        __syncthreads();
        #pragma unroll
        for (int i = 0; i < 4; ++i) {
            *(v8h*)(Asw + 32 * i * LDSP) = *(const v8h*)(Ag + (size_t)(32 * i) * lda + kt);
            *(v8h*)(Bsw + 32 * i * LDSP) = *(const v8h*)(Bg + (size_t)(32 * i) * ldb + kt);
        }
        __syncthreads();
        #pragma unroll
        for (int ks = 0; ks < 64; ks += 32) {
            int kb = ks + kqh;
            v16h a0  = *(const v16h*)(As + (wm * 32      + lr) * LDSP + kb);
            v16h a1  = *(const v16h*)(As + (wm * 32 + 16 + lr) * LDSP + kb);
            v16h b0  = *(const v16h*)(Bs + (wn * 64      + lr) * LDSP + kb);
            v16h b1  = *(const v16h*)(Bs + (wn * 64 + 16 + lr) * LDSP + kb);
            v16h b2v = *(const v16h*)(Bs + (wn * 64 + 32 + lr) * LDSP + kb);
            v16h b3v = *(const v16h*)(Bs + (wn * 64 + 48 + lr) * LDSP + kb);
            acc[0][0] = WMMA_F16(a0, b0,  acc[0][0]);
            acc[0][1] = WMMA_F16(a0, b1,  acc[0][1]);
            acc[0][2] = WMMA_F16(a0, b2v, acc[0][2]);
            acc[0][3] = WMMA_F16(a0, b3v, acc[0][3]);
            acc[1][0] = WMMA_F16(a1, b0,  acc[1][0]);
            acc[1][1] = WMMA_F16(a1, b1,  acc[1][1]);
            acc[1][2] = WMMA_F16(a1, b2v, acc[1][2]);
            acc[1][3] = WMMA_F16(a1, b3v, acc[1][3]);
        }
    }
#endif
#endif

    // C layout: element (m, n) -> vgpr m&7, lane (n&15) + 16*(m>>3)
    int mo = (lane >> 4) * 8;
    int mW = blockIdx.y * 128 + wm * 32;
    int nW = blockIdx.x * 128 + wn * 64;
    #pragma unroll
    for (int mi = 0; mi < 2; ++mi) {
        #pragma unroll
        for (int nj = 0; nj < 4; ++nj) {
            #pragma unroll
            for (int r = 0; r < 8; ++r) {
                int gm = mW + mi * 16 + mo + r;
                int gn = nW + nj * 16 + lr;
                float v = acc[mi][nj][r];
                if (EPI == 0) {
                    ((float*)C)[Coff + (size_t)gm * ldc + gn] = v;
                } else if (EPI == 4) {
                    __builtin_nontemporal_store(v, &((float*)C)[Coff + (size_t)gm * ldc + gn]);
                } else if (EPI == 1) {
                    ((_Float16*)C)[Coff + (size_t)gm * ldc + gn] = (_Float16)v;
                } else if (EPI == 2) {
                    v += biasb[gn];
                    v = 0.5f * v * (1.0f + erff(v * 0.70710678118654752f));
                    ((_Float16*)C)[Coff + (size_t)gm * ldc + gn] = (_Float16)v;
                } else { // EPI == 3: bias, store transposed C^T[n][m]
                    v += biasb[gn];
                    ((_Float16*)C)[Coff + (size_t)gn * ldc + gm] = (_Float16)v;
                }
            }
        }
    }
}

extern "C" void kernel_launch(void* const* d_in, const int* in_sizes, int n_in,
                              void* d_out, int out_size, void* d_ws, size_t ws_size,
                              hipStream_t stream) {
    const float* x     = (const float*)d_in[0];
    const float* gamma = (const float*)d_in[1];
    const float* beta  = (const float*)d_in[2];
    const float* mu    = (const float*)d_in[3];
    const float* scale = (const float*)d_in[4];
    const float* W1    = (const float*)d_in[5];
    const float* b1    = (const float*)d_in[6];
    const float* W2    = (const float*)d_in[7];
    const float* b2    = (const float*)d_in[8];
    float* out = (float*)d_out;

    // workspace carve-up (256B aligned)
    char* p = (char*)d_ws;
    auto alloc = [&](size_t bytes) { void* r = (void*)p; p += (bytes + 255) & ~(size_t)255; return r; };
    _Float16* xln    = (_Float16*)alloc((size_t)BN_ * D_ * 2);      // [BN][D]
    _Float16* xn     = (_Float16*)alloc((size_t)BN_ * D_ * 2);      // [BN][D]
    _Float16* xlnT   = (_Float16*)alloc((size_t)B_ * D_ * N_ * 2);  // [b][D][N]
    _Float16* munT   = (_Float16*)alloc((size_t)ES_ * D_ * 2);      // [ES][D]
    float*    logits = (float*)   alloc((size_t)BN_ * ES_ * 4);     // [b][n][es]
    _Float16* dispT  = (_Float16*)alloc((size_t)B_ * ES_ * N_ * 2); // [b][es][n]
    _Float16* comb   = (_Float16*)alloc((size_t)BN_ * ES_ * 2);     // [b][n][es]
    _Float16* slotin = (_Float16*)alloc((size_t)B_ * ES_ * D_ * 2); // [b][es][d]
    _Float16* w1T    = (_Float16*)alloc((size_t)E_ * H_ * D_ * 2);  // [e][H][D]
    _Float16* w2T    = (_Float16*)alloc((size_t)E_ * D_ * H_ * 2);  // [e][D][H]
    _Float16* hbuf   = (_Float16*)alloc((size_t)B_ * ES_ * H_ * 2); // [b][es][h]
    _Float16* soT    = (_Float16*)alloc((size_t)B_ * D_ * ES_ * 2); // [b][d][es]

    dim3 t256(256), t32x8(32, 8);

    // weight transposes (fp32 -> f16, col-major operands for WMMA B)
    transpose_to_f16<float><<<dim3(H_ / 32, D_ / 32, E_), t32x8, 0, stream>>>(W1, w1T, D_, H_);
    transpose_to_f16<float><<<dim3(D_ / 32, H_ / 32, E_), t32x8, 0, stream>>>(W2, w2T, H_, D_);
    mu_norm_kernel<<<dim3(ES_ / 256), t256, 0, stream>>>(mu, munT);

    // LayerNorm + l2norm
    ln_kernel<<<dim3(BN_), t256, 0, stream>>>(x, gamma, beta, scale, xln, xn);
    transpose_to_f16<_Float16><<<dim3(D_ / 32, N_ / 32, B_), t32x8, 0, stream>>>(xln, xlnT, N_, D_);

    // GEMM-1: logits[BN][ES] = xn[BN][D] * munT[ES][D]^T
    gemm_wmma_f16<0><<<dim3(ES_ / 128, BN_ / 128, 1), t256, 0, stream>>>(
        xn, munT, logits, nullptr, D_, D_, D_, ES_,
        0LL, 0LL, 1, 0LL, 0LL, 1, 0LL);

    // softmaxes
    dispatch_softmax_kernel<<<dim3(ES_ / 256, B_), t256, 0, stream>>>(logits, dispT);
    combine_softmax_kernel<<<dim3(BN_), t256, 0, stream>>>(logits, comb);

    // GEMM-2: slot_in[b][ES][D] = dispT[b][ES][N] * xlnT[b][D][N]^T
    gemm_wmma_f16<1><<<dim3(D_ / 128, ES_ / 128, B_), t256, 0, stream>>>(
        dispT, xlnT, slotin, nullptr, N_, N_, N_, D_,
        (long long)ES_ * N_, (long long)D_ * N_, B_,
        (long long)ES_ * D_, 0LL, 1, 0LL);

    // GEMM-3: h[b][e][S][H] = gelu(slot_in[b][e][S][D] * w1T[e][H][D]^T + b1[e])
    gemm_wmma_f16<2><<<dim3(H_ / 128, S_ / 128, B_ * E_), t256, 0, stream>>>(
        slotin, w1T, hbuf, b1, D_, D_, D_, H_,
        (long long)S_ * D_, (long long)H_ * D_, E_,
        (long long)S_ * H_, 0LL, 1, (long long)H_);

    // GEMM-4: soT[b][d][e*S+s] = (h[b][e][S][H] * w2T[e][D][H]^T + b2[e]) stored transposed
    gemm_wmma_f16<3><<<dim3(D_ / 128, S_ / 128, B_ * E_), t256, 0, stream>>>(
        hbuf, w2T, soT, b2, H_, H_, H_, ES_,
        (long long)S_ * H_, (long long)D_ * H_, E_,
        (long long)D_ * ES_, (long long)S_, E_, (long long)D_);

    // GEMM-5: out[b][N][D] = comb[b][N][ES] * soT[b][D][ES]^T  (fp32, non-temporal)
    gemm_wmma_f16<4><<<dim3(D_ / 128, N_ / 128, B_), t256, 0, stream>>>(
        comb, soT, out, nullptr, ES_, ES_, ES_, D_,
        (long long)N_ * ES_, (long long)D_ * ES_, B_,
        (long long)N_ * D_, 0LL, 1, 0LL);
}
